// Attention_16106127360635
// MI455X (gfx1250) — matrix operand order repaired and checked by reading, not compiled, not run on hardware
//
#include <hip/hip_runtime.h>
#include <hip/hip_bf16.h>

typedef _Float16 f16;
typedef __attribute__((ext_vector_type(8)))  _Float16 v8h;
typedef __attribute__((ext_vector_type(16))) _Float16 v16h;
typedef __attribute__((ext_vector_type(8)))  float    v8f;

#define BATCH 8
#define SEQ   2048
#define NSENT 512
#define EMB   1024

__device__ __forceinline__ v8f wmma16(v16h a, v16h b, v8f c) {
    // D = A(16x32 f16) * B(32x16 f16) + C(16x16 f32)
    return __builtin_amdgcn_wmma_f32_16x16x32_f16(
        /*neg_a=*/false, a, /*neg_b=*/false, b,
        /*c_mod=*/(short)0, c, /*reuse_a=*/false, /*reuse_b=*/false);
}

// A fragment (16x32 f16): rows m0..m0+15, K = kb..kb+31, row-major src, leading dim ld.
// Layout: lanes 0-15 hold row (lane), K {0..15}; lanes 16-31 K {16..31}.
__device__ __forceinline__ v16h load_a(const f16* base, int ld, int m0, int kb, int lane) {
    int row   = m0 + (lane & 15);
    int khalf = (lane >> 4) << 4;            // 0 or 16
    const f16* p = base + (size_t)row * ld + kb + khalf;
    v8h lo = *(const v8h*)p;                 // K kb+khalf   .. +7
    v8h hi = *(const v8h*)(p + 8);           // K kb+khalf+8 .. +15
    return __builtin_shufflevector(lo, hi, 0,1,2,3,4,5,6,7,8,9,10,11,12,13,14,15);
}

// B fragment (32x16 f16): cols n0..n0+15, K = kb..kb+31. Memory holds B^T row-major
// (column n of B is a contiguous K-run). Lane holds column (lane&15), 16 consecutive
// K values; lanes 16-31 take K+16.
__device__ __forceinline__ v16h load_bt(const f16* base, int ld, int n0, int kb, int lane) {
    int col = n0 + (lane & 15);
    int k2  = kb + ((lane >> 4) << 4);       // 0 or +16
    return *(const v16h*)(base + (size_t)col * ld + k2);
}

// Async global->LDS copy of 16 bytes (gfx1250 GLOBAL_LOAD_ASYNC_TO_LDS_B128, ASYNCcnt).
__device__ __forceinline__ void async_g2l_b128(unsigned lds_addr, const void* gptr) {
    asm volatile("global_load_async_to_lds_b128 %0, %1, off"
                 :: "v"(lds_addr), "v"((unsigned long long)(uintptr_t)gptr)
                 : "memory");
}
__device__ __forceinline__ void wait_async0() {
    asm volatile("s_wait_asynccnt 0x0" ::: "memory");
}

// ---------------- conversion kernels ----------------
__global__ void cvt_f32_to_f16(const float* __restrict__ in, f16* __restrict__ out, int n) {
    int i = blockIdx.x * blockDim.x + threadIdx.x;
    int stride = gridDim.x * blockDim.x;
    for (; i < n; i += stride) out[i] = (f16)in[i];
}

// sentence (B, N, E) f32 -> ShT (B, E, N) f16, LDS-tiled 32x32 transpose
__global__ void transpose_sent(const float* __restrict__ in, f16* __restrict__ out) {
    __shared__ f16 tile[32][33];
    int b  = blockIdx.z;
    int e0 = blockIdx.x * 32;
    int n0 = blockIdx.y * 32;
    int tx = threadIdx.x, ty = threadIdx.y;   // block (32, 8)
    #pragma unroll
    for (int j = ty; j < 32; j += 8)
        tile[j][tx] = (f16)in[((size_t)b * NSENT + (n0 + j)) * EMB + e0 + tx];
    __syncthreads();
    #pragma unroll
    for (int j = ty; j < 32; j += 8)
        out[((size_t)b * EMB + (e0 + j)) * NSENT + n0 + tx] = tile[tx][j];
}

// ---------------- projection GEMM: w = word @ W^T ----------------
// Each wave computes a 16x128 strip; per k-step: 1 A frag + 8 B frags batched,
// then 8 back-to-back WMMAs (one loadcnt wait per k-step instead of per WMMA).
__global__ void proj_gemm(const f16* __restrict__ A, const f16* __restrict__ Wh,
                          float* __restrict__ comb, f16* __restrict__ wh16) {
    int wid  = (blockIdx.x * blockDim.x + threadIdx.x) >> 5;
    int lane = threadIdx.x & 31;
    int nt = wid & 7;          // 8 strips of 128 cols
    int mt = wid >> 3;         // 1024 row tiles
    int m0 = mt * 16;
    int n0 = nt * 128;

    v8f acc[8];
    #pragma unroll
    for (int j = 0; j < 8; j++) acc[j] = v8f{0.f,0.f,0.f,0.f,0.f,0.f,0.f,0.f};

    for (int kb = 0; kb < EMB; kb += 32) {
        v16h a = load_a(A, EMB, m0, kb, lane);
        v16h bfr[8];
        #pragma unroll
        for (int j = 0; j < 8; j++) bfr[j] = load_bt(Wh, EMB, n0 + j * 16, kb, lane);
        #pragma unroll
        for (int j = 0; j < 8; j++) acc[j] = wmma16(a, bfr[j], acc[j]);
    }

    int rr = (lane >> 4) << 3;
    int cc = lane & 15;
    #pragma unroll
    for (int j = 0; j < 8; j++) {
        int col = n0 + j * 16 + cc;
        #pragma unroll
        for (int r = 0; r < 8; r++) {
            int row = m0 + rr + r;                 // row == b*SEQ + s
            float v = acc[j][r];
            comb[(size_t)row * (2 * EMB) + col] = v;   // combination[..., :E]
            wh16[(size_t)row * EMB + col] = (f16)v;    // f16 copy for attention
        }
    }
}

// ---------------- attention: scores -> softmax -> g ----------------
// Two waves per (b, 16-row s-tile); block = 64. grid = 8 * 128 = 1024 blocks.
// LDS (64 KB): [0,32K) scores f32; [32K,64K) w-tile f16, later overlaid by
// att f16 (16 KB) + softmax partials.
__global__ void __launch_bounds__(64)
attn_kernel(const f16* __restrict__ wh16, const f16* __restrict__ Sh,
            const f16* __restrict__ ShT,
            float* __restrict__ comb, float* __restrict__ gout) {
    __shared__ __align__(16) char smem[65536];
    float* sc    = (float*)smem;                      // 16 x 512 f32
    f16*   wtile = (f16*)(smem + 32768);              // 16 x 1024 f16 (phase 1 only)
    f16*   attv  = (f16*)(smem + 32768);              // 16 x 512 f16 (phase >= 2)
    float* pmax  = (float*)(smem + 32768 + 16384);    // 16 x 4
    float* psum  = pmax + 64;                         // 16 x 4

    int tid  = threadIdx.x;       // 0..63
    int lane = tid & 31;
    int wave = tid >> 5;          // 0 or 1
    int tile = blockIdx.x;
    int b  = tile >> 7;           // 128 s-tiles per batch
    int s0 = (tile & 127) * 16;

    const f16* Wrow = wh16 + (size_t)b * SEQ * EMB;
    const f16* Srow = Sh   + (size_t)b * NSENT * EMB;
    const f16* ST   = ShT  + (size_t)b * EMB * NSENT;

    int rr = (lane >> 4) << 3;
    int cc = lane & 15;

    // ---- stage w tile (16 rows = contiguous 32 KB) into LDS via async copy ----
    {
        const char* src = (const char*)(Wrow + (size_t)s0 * EMB);
        unsigned ldsbase = (unsigned)(uintptr_t)wtile;
        #pragma unroll
        for (int i = 0; i < 32; i++) {
            unsigned off = (unsigned)(i * 64 + tid) * 16;   // 64 thr * 16 B * 32 it = 32 KB
            async_g2l_b128(ldsbase + off, src + off);
        }
        wait_async0();
    }
    __syncthreads();

    // ---- Phase 1: scores[16][512] = w_tile(16xE) @ sent^T; wave w owns 256 cols ----
    for (int g = 0; g < 4; g++) {
        int nb0 = wave * 256 + g * 64;
        v8f acc[4];
        #pragma unroll
        for (int j = 0; j < 4; j++) acc[j] = v8f{0.f,0.f,0.f,0.f,0.f,0.f,0.f,0.f};
        for (int kb = 0; kb < EMB; kb += 32) {
            v16h a = load_a(wtile, EMB, 0, kb, lane);       // from LDS
            v16h bfr[4];
            #pragma unroll
            for (int j = 0; j < 4; j++) bfr[j] = load_bt(Srow, EMB, nb0 + j * 16, kb, lane);
            #pragma unroll
            for (int j = 0; j < 4; j++) acc[j] = wmma16(a, bfr[j], acc[j]);
        }
        #pragma unroll
        for (int j = 0; j < 4; j++)
            #pragma unroll
            for (int r = 0; r < 8; r++)
                sc[(rr + r) * NSENT + nb0 + j * 16 + cc] = acc[j][r];
    }
    __syncthreads();

    // ---- Phase 2: softmax per row; 4 threads per row, 128 cols each ----
    {
        int row = tid & 15;
        int q   = tid >> 4;                    // 0..3
        const float* prow = sc + row * NSENT + q * 128;
        float m = -3.402823e38f;
        for (int i = 0; i < 128; i++) m = fmaxf(m, prow[i]);
        pmax[row * 4 + q] = m;
        __syncthreads();
        float M = fmaxf(fmaxf(pmax[row * 4 + 0], pmax[row * 4 + 1]),
                        fmaxf(pmax[row * 4 + 2], pmax[row * 4 + 3]));
        float s = 0.f;
        f16* arow = attv + row * NSENT + q * 128;     // overlays dead w-tile region
        for (int i = 0; i < 128; i++) {
            float e = __expf(prow[i] - M);
            s += e;
            arow[i] = (f16)e;
        }
        psum[row * 4 + q] = s;
        __syncthreads();
        float S = (psum[row * 4 + 0] + psum[row * 4 + 1]) +
                  (psum[row * 4 + 2] + psum[row * 4 + 3]);
        float inv = 1.0f / S;
        for (int i = 0; i < 128; i++)
            arow[i] = (f16)((float)arow[i] * inv);
    }
    __syncthreads();

    // ---- Phase 3: g[16][1024] = att(16x512) @ sent; wave w owns 512 e-cols ----
    for (int g = 0; g < 8; g++) {
        int et0 = wave * 512 + g * 64;
        v8f acc[4];
        #pragma unroll
        for (int j = 0; j < 4; j++) acc[j] = v8f{0.f,0.f,0.f,0.f,0.f,0.f,0.f,0.f};
        for (int kb = 0; kb < NSENT; kb += 32) {
            v16h a = load_a(attv, NSENT, 0, kb, lane);      // from LDS
            v16h bfr[4];
            #pragma unroll
            for (int j = 0; j < 4; j++) bfr[j] = load_bt(ST, NSENT, et0 + j * 16, kb, lane);
            #pragma unroll
            for (int j = 0; j < 4; j++) acc[j] = wmma16(a, bfr[j], acc[j]);
        }
        #pragma unroll
        for (int j = 0; j < 4; j++) {
            #pragma unroll
            for (int r = 0; r < 8; r++) {
                int s = s0 + rr + r;
                int e = et0 + j * 16 + cc;
                float v = acc[j][r];
                size_t row = (size_t)b * SEQ + s;
                comb[row * (2 * EMB) + EMB + e] = v;   // combination[..., E:]
                gout[row * EMB + e] = v;               // g output
            }
        }
    }
}

extern "C" void kernel_launch(void* const* d_in, const int* in_sizes, int n_in,
                              void* d_out, int out_size, void* d_ws, size_t ws_size,
                              hipStream_t stream) {
    const float* word = (const float*)d_in[0];   // (8, 2048, 1024)
    const float* sent = (const float*)d_in[1];   // (8, 512, 1024)
    const float* Wf   = (const float*)d_in[2];   // (1024, 1024)

    const size_t nWord = (size_t)BATCH * SEQ * EMB;     // 16777216
    const size_t nW    = (size_t)EMB * EMB;             // 1048576
    const size_t nSent = (size_t)BATCH * NSENT * EMB;   // 4194304

    f16* Ah   = (f16*)d_ws;          // word  f16
    f16* Wh   = Ah + nWord;          // W     f16
    f16* Sh   = Wh + nW;             // sent  f16 (N-major)
    f16* ShT  = Sh + nSent;          // sent  f16 (E-major, transposed)
    f16* wh16 = ShT + nSent;         // w     f16

    float* comb = (float*)d_out;                               // (8,2048,2048)
    float* gout = comb + (size_t)BATCH * SEQ * 2 * EMB;        // (8,2048,1024)

    cvt_f32_to_f16<<<(int)((nWord + 1023) / 1024), 256, 0, stream>>>(word, Ah, (int)nWord);
    cvt_f32_to_f16<<<(int)((nW    + 1023) / 1024), 256, 0, stream>>>(Wf,   Wh, (int)nW);
    cvt_f32_to_f16<<<(int)((nSent + 1023) / 1024), 256, 0, stream>>>(sent, Sh, (int)nSent);

    dim3 tgrid(EMB / 32, NSENT / 32, BATCH);
    transpose_sent<<<tgrid, dim3(32, 8), 0, stream>>>(sent, ShT);

    // 8192 waves, 8 waves per block -> 1024 blocks
    proj_gemm<<<1024, 256, 0, stream>>>(Ah, Wh, comb, wh16);

    // two waves per 16-row s-tile: 8 * 128 = 1024 blocks of 64 threads
    attn_kernel<<<BATCH * (SEQ / 16), 64, 0, stream>>>(wh16, Sh, ShT, comb, gout);
}
